// GATModel_38242388804051
// MI455X (gfx1250) — matrix operand order repaired and checked by reading, not compiled, not run on hardware
//
#include <hip/hip_runtime.h>

#define N_NODES 50000
#define N_EDGES 800000
#define E_TOT   (N_NODES + N_EDGES)
#define NGRP    64

typedef __attribute__((ext_vector_type(16))) _Float16 v16h;
typedef __attribute__((ext_vector_type(8)))  float    v8f;

// ---------- helpers: order-preserving float<->uint map for atomic max ----------
__device__ __forceinline__ unsigned ord_map(float f) {
  unsigned u = __float_as_uint(f);
  return (u & 0x80000000u) ? ~u : (u | 0x80000000u);
}
__device__ __forceinline__ float ord_unmap(unsigned u) {
  return __uint_as_float((u & 0x80000000u) ? (u & 0x7FFFFFFFu) : ~u);
}
// ord_map(-inf) == 0x007FFFFFu

// ---------- generic fill ----------
__global__ void fill_u32(unsigned* __restrict__ p, unsigned val, int n) {
  int i = blockIdx.x * blockDim.x + threadIdx.x;
  if (i < n) p[i] = val;
}

// ---------- pack W (fp32 [128,Ncols]) into f16 WMMA B-fragment order ----------
// Bp[((ks*colTiles + ct)*32 + lane)*16 + j] = (f16) W[k*Ncols + n]
//   with k = ks*32 + (j>=8)*16 + (lane>=16)*8 + (j&7) ,  n = ct*16 + (lane&15)
__global__ void pack_w_kernel(const float* __restrict__ W, _Float16* __restrict__ Bp, int Ncols) {
  int idx = blockIdx.x * blockDim.x + threadIdx.x;
  if (idx >= 128 * Ncols) return;
  int colTiles = Ncols >> 4;
  int j  = idx & 15;
  int L  = (idx >> 4) & 31;
  int rest = idx >> 9;
  int ct = rest % colTiles;
  int ks = rest / colTiles;
  int k = (ks << 5) + ((j >> 3) << 4) + ((L >> 4) << 3) + (j & 7);
  int n = (ct << 4) + (L & 15);
  Bp[idx] = (_Float16)W[k * Ncols + n];
}

// ---------- WMMA GEMM: C[M,Ncols] = A[M,128] * W[128,Ncols] ----------
// One wave owns one 16-row strip; A fragments loaded once (b128 loads, packed cvt),
// reused across all column tiles; B fragments are contiguous 32B v16h loads.
__global__ void gemm_wmma_kernel(const float* __restrict__ A, const _Float16* __restrict__ Bp,
                                 float* __restrict__ Cout, int M, int Ncols) {
  const int lane = threadIdx.x;                       // 0..31 (wave32)
  const int colTiles = Ncols >> 4;
  int rowTile = blockIdx.x * blockDim.y + threadIdx.y;
  if ((rowTile << 4) >= M) return;                    // wave-uniform exit; M % 16 == 0
  const int m  = lane & 15;
  const int hi = lane >> 4;
  const float* arow = A + (size_t)((rowTile << 4) + m) * 128 + (hi << 3);

  v16h afrag[4];
#pragma unroll
  for (int ks = 0; ks < 4; ++ks) {
    float4 l0 = *(const float4*)(arow + ks * 32);
    float4 l1 = *(const float4*)(arow + ks * 32 + 4);
    float4 h0 = *(const float4*)(arow + ks * 32 + 16);
    float4 h1 = *(const float4*)(arow + ks * 32 + 20);
    v16h a;
    a[0]  = (_Float16)l0.x; a[1]  = (_Float16)l0.y; a[2]  = (_Float16)l0.z; a[3]  = (_Float16)l0.w;
    a[4]  = (_Float16)l1.x; a[5]  = (_Float16)l1.y; a[6]  = (_Float16)l1.z; a[7]  = (_Float16)l1.w;
    a[8]  = (_Float16)h0.x; a[9]  = (_Float16)h0.y; a[10] = (_Float16)h0.z; a[11] = (_Float16)h0.w;
    a[12] = (_Float16)h1.x; a[13] = (_Float16)h1.y; a[14] = (_Float16)h1.z; a[15] = (_Float16)h1.w;
    afrag[ks] = a;
  }

#pragma unroll
  for (int ct = 0; ct < 8; ++ct) {
    if (ct >= colTiles) break;                        // uniform
    v8f acc = {};
#pragma unroll
    for (int ks = 0; ks < 4; ++ks) {
      v16h b = *(const v16h*)(Bp + (size_t)(((ks * colTiles + ct) << 5) + lane) * 16);
      acc = __builtin_amdgcn_wmma_f32_16x16x32_f16(
          false, afrag[ks], false, b, (short)0, acc, false, false);
    }
    float* cp = Cout + (size_t)((rowTile << 4) + (hi << 3)) * Ncols + (ct << 4) + m;
#pragma unroll
    for (int r = 0; r < 8; ++r) cp[(size_t)r * Ncols] = acc[r];
  }
}

// ---------- attention logits: al[n,h] = sum_c h[n,h,c]*a[h,c] ----------
__global__ void attn_kernel(const float* __restrict__ h, const float* __restrict__ a_s,
                            const float* __restrict__ a_d, float* __restrict__ als,
                            float* __restrict__ ald, int n_, int H, int C) {
  int idx = blockIdx.x * blockDim.x + threadIdx.x;
  if (idx >= n_ * H) return;
  int node = idx / H, head = idx - node * H;
  const float* hp = h + (size_t)node * H * C + head * C;
  const float* as = a_s + head * C;
  const float* ad = a_d + head * C;
  float ss = 0.f, sd = 0.f;
  for (int c = 0; c < C; ++c) { float v = hp[c]; ss += v * as[c]; sd += v * ad[c]; }
  als[idx] = ss; ald[idx] = sd;
}

// ---------- pass 1: e = leaky_relu(al_s[src]+al_d[dst]); segment max via atomicMax ----------
__global__ void edge_logits_max(const int* __restrict__ ei, int E_, int Etot,
                                const float* __restrict__ als, const float* __restrict__ ald,
                                float* __restrict__ ebuf, unsigned* __restrict__ mbuf, int H) {
  int idx = blockIdx.x * blockDim.x + threadIdx.x;
  if (idx >= Etot * H) return;
  int e = idx / H, head = idx - e * H;
  int s, d;
  if (e < E_) { s = ei[e]; d = ei[E_ + e]; } else { s = d = e - E_; }   // self-loops
  float v = als[s * H + head] + ald[d * H + head];
  v = v > 0.f ? v : 0.2f * v;                                          // leaky_relu(0.2)
  ebuf[idx] = v;
  atomicMax(&mbuf[d * H + head], ord_map(v));
}

// ---------- pass 2: ex = exp(e - m[dst]); segment sum ----------
__global__ void edge_exp_sum(const int* __restrict__ ei, int E_, int Etot,
                             const unsigned* __restrict__ mbuf, float* __restrict__ ebuf,
                             float* __restrict__ sbuf, int H) {
  int idx = blockIdx.x * blockDim.x + threadIdx.x;
  if (idx >= Etot * H) return;
  int e = idx / H, head = idx - e * H;
  int d = (e < E_) ? ei[E_ + e] : (e - E_);
  float mx = ord_unmap(mbuf[d * H + head]);
  float ex = expf(ebuf[idx] - mx);
  ebuf[idx] = ex;
  atomicAdd(&sbuf[d * H + head], ex);
}

// ---------- pass 2b: alpha = ex / (s[dst] + eps) (in place) ----------
__global__ void edge_alpha(const int* __restrict__ ei, int E_, int Etot,
                           float* __restrict__ ebuf, const float* __restrict__ sbuf, int H) {
  int idx = blockIdx.x * blockDim.x + threadIdx.x;
  if (idx >= Etot * H) return;
  int e = idx / H, head = idx - e * H;
  int d = (e < E_) ? ei[E_ + e] : (e - E_);
  ebuf[idx] = ebuf[idx] / (sbuf[d * H + head] + 1e-16f);
}

// ---------- pass 3: out[dst] += h[src] * alpha  (thread per (edge, feature)) ----------
__global__ void edge_aggregate(const int* __restrict__ ei, int E_, int Etot,
                               const float* __restrict__ hsrc, const float* __restrict__ ebuf,
                               float* __restrict__ out, int H, int C) {
  int F = H * C;
  int idx = blockIdx.x * blockDim.x + threadIdx.x;
  if (idx >= Etot * F) return;
  int e = idx / F, f = idx - e * F;
  int head = f / C;
  int s, d;
  if (e < E_) { s = ei[e]; d = ei[E_ + e]; } else { s = d = e - E_; }
  float alpha = ebuf[e * H + head];
  atomicAdd(&out[(size_t)d * F + f], hsrc[(size_t)s * F + f] * alpha);
}

// ---------- batchnorm stats (F==128): block of 128 threads, strided rows ----------
__global__ void bn_stats(const float* __restrict__ x, float* __restrict__ sum,
                         float* __restrict__ sumsq, int n_, int F) {
  int f = threadIdx.x;
  float s = 0.f, q = 0.f;
  for (int n = blockIdx.x; n < n_; n += gridDim.x) {
    float v = x[(size_t)n * F + f]; s += v; q += v * v;
  }
  atomicAdd(&sum[f], s);
  atomicAdd(&sumsq[f], q);
}

// ---------- batchnorm apply + ELU (in place) ----------
__global__ void bn_apply(float* __restrict__ x, const float* __restrict__ sum,
                         const float* __restrict__ sumsq, const float* __restrict__ g,
                         const float* __restrict__ be, int n_, int F) {
  int idx = blockIdx.x * blockDim.x + threadIdx.x;
  if (idx >= n_ * F) return;
  int f = idx % F;
  float inv_n = 1.0f / (float)n_;
  float mu  = sum[f] * inv_n;
  float var = sumsq[f] * inv_n - mu * mu;
  float v = (x[idx] - mu) * rsqrtf(var + 1e-5f) * g[f] + be[f];
  x[idx] = v > 0.f ? v : (expf(v) - 1.f);      // ELU
}

// ---------- global mean pool ----------
__global__ void pool_accum(const float* __restrict__ x, const int* __restrict__ batch,
                           const float* __restrict__ b3, float* __restrict__ outsum,
                           float* __restrict__ cnt, int n_, int F) {
  int idx = blockIdx.x * blockDim.x + threadIdx.x;
  if (idx >= n_ * F) return;
  int n = idx / F, f = idx - n * F;
  int g = batch[n];
  atomicAdd(&outsum[g * F + f], x[idx] + b3[f]);
  if (f == 0) atomicAdd(&cnt[g], 1.0f);
}
__global__ void pool_final(float* __restrict__ out, const float* __restrict__ cnt, int ng, int F) {
  int idx = blockIdx.x * blockDim.x + threadIdx.x;
  if (idx >= ng * F) return;
  out[idx] /= fmaxf(cnt[idx / F], 1.0f);
}

// =====================================================================
extern "C" void kernel_launch(void* const* d_in, const int* in_sizes, int n_in,
                              void* d_out, int out_size, void* d_ws, size_t ws_size,
                              hipStream_t stream) {
  const float* x     = (const float*)d_in[0];
  const int*   ei    = (const int*)d_in[1];    // [2,E] int32
  const int*   batch = (const int*)d_in[2];    // [N]  int32
  const float* Wm[4] = {(const float*)d_in[3],  (const float*)d_in[7],
                        (const float*)d_in[11], (const float*)d_in[15]};
  const float* As[4] = {(const float*)d_in[4],  (const float*)d_in[8],
                        (const float*)d_in[12], (const float*)d_in[16]};
  const float* Ad[4] = {(const float*)d_in[5],  (const float*)d_in[9],
                        (const float*)d_in[13], (const float*)d_in[17]};
  const float* Bs3   = (const float*)d_in[18];
  const float* Gg[3] = {(const float*)d_in[19], (const float*)d_in[21], (const float*)d_in[23]};
  const float* Be[3] = {(const float*)d_in[20], (const float*)d_in[22], (const float*)d_in[24]};

  float* ws = (float*)d_ws;
  size_t o = 0;
  float*    hA    = ws + o; o += (size_t)N_NODES * 128;
  float*    hB    = ws + o; o += (size_t)N_NODES * 128;
  float*    als   = ws + o; o += (size_t)N_NODES * 4;
  float*    ald   = ws + o; o += (size_t)N_NODES * 4;
  unsigned* mbuf  = (unsigned*)(ws + o); o += (size_t)N_NODES * 4;
  float*    sbuf  = ws + o; o += (size_t)N_NODES * 4;
  float*    ebuf  = ws + o; o += (size_t)E_TOT * 4;
  float*    bnsum = ws + o; o += 128;
  float*    bnsq  = ws + o; o += 128;
  float*    cnt   = ws + o; o += NGRP;
  _Float16* wpack = (_Float16*)(ws + o); o += 128 * 128 / 2;   // f16 W fragments (32KB)

  auto cdiv = [](long long a, long long b) { return (unsigned)((a + b - 1) / b); };
  const unsigned rowTiles = N_NODES / 16;                       // 3125 (exact)

  const float* in = x;
  // ----- layers 0..2: H=4, C=32, F=128 -----
  for (int l = 0; l < 3; ++l) {
    const int H = 4, C = 32, F = 128;
    pack_w_kernel<<<cdiv(128 * F, 256), 256, 0, stream>>>(Wm[l], wpack, F);
    gemm_wmma_kernel<<<cdiv(rowTiles, 8), dim3(32, 8), 0, stream>>>(in, wpack, hB, N_NODES, F);
    attn_kernel<<<cdiv((long long)N_NODES * H, 256), 256, 0, stream>>>(hB, As[l], Ad[l], als, ald, N_NODES, H, C);
    fill_u32<<<cdiv((long long)N_NODES * H, 256), 256, 0, stream>>>(mbuf, 0x007FFFFFu, N_NODES * H);
    fill_u32<<<cdiv((long long)N_NODES * H, 256), 256, 0, stream>>>((unsigned*)sbuf, 0u, N_NODES * H);
    fill_u32<<<cdiv((long long)N_NODES * F, 256), 256, 0, stream>>>((unsigned*)hA, 0u, N_NODES * F);
    edge_logits_max<<<cdiv((long long)E_TOT * H, 256), 256, 0, stream>>>(ei, N_EDGES, E_TOT, als, ald, ebuf, mbuf, H);
    edge_exp_sum<<<cdiv((long long)E_TOT * H, 256), 256, 0, stream>>>(ei, N_EDGES, E_TOT, mbuf, ebuf, sbuf, H);
    edge_alpha<<<cdiv((long long)E_TOT * H, 256), 256, 0, stream>>>(ei, N_EDGES, E_TOT, ebuf, sbuf, H);
    edge_aggregate<<<cdiv((long long)E_TOT * F, 256), 256, 0, stream>>>(ei, N_EDGES, E_TOT, hB, ebuf, hA, H, C);
    fill_u32<<<1, 256, 0, stream>>>((unsigned*)bnsum, 0u, 128);
    fill_u32<<<1, 256, 0, stream>>>((unsigned*)bnsq, 0u, 128);
    bn_stats<<<512, 128, 0, stream>>>(hA, bnsum, bnsq, N_NODES, F);
    bn_apply<<<cdiv((long long)N_NODES * F, 256), 256, 0, stream>>>(hA, bnsum, bnsq, Gg[l], Be[l], N_NODES, F);
    in = hA;
  }
  // ----- final layer: H=1, C=64, F=64 -----
  {
    const int H = 1, C = 64, F = 64;
    pack_w_kernel<<<cdiv(128 * F, 256), 256, 0, stream>>>(Wm[3], wpack, F);
    gemm_wmma_kernel<<<cdiv(rowTiles, 8), dim3(32, 8), 0, stream>>>(in, wpack, hB, N_NODES, F);
    attn_kernel<<<cdiv((long long)N_NODES * H, 256), 256, 0, stream>>>(hB, As[3], Ad[3], als, ald, N_NODES, H, C);
    fill_u32<<<cdiv((long long)N_NODES * H, 256), 256, 0, stream>>>(mbuf, 0x007FFFFFu, N_NODES * H);
    fill_u32<<<cdiv((long long)N_NODES * H, 256), 256, 0, stream>>>((unsigned*)sbuf, 0u, N_NODES * H);
    fill_u32<<<cdiv((long long)N_NODES * F, 256), 256, 0, stream>>>((unsigned*)hA, 0u, N_NODES * F);
    edge_logits_max<<<cdiv((long long)E_TOT * H, 256), 256, 0, stream>>>(ei, N_EDGES, E_TOT, als, ald, ebuf, mbuf, H);
    edge_exp_sum<<<cdiv((long long)E_TOT * H, 256), 256, 0, stream>>>(ei, N_EDGES, E_TOT, mbuf, ebuf, sbuf, H);
    edge_alpha<<<cdiv((long long)E_TOT * H, 256), 256, 0, stream>>>(ei, N_EDGES, E_TOT, ebuf, sbuf, H);
    edge_aggregate<<<cdiv((long long)E_TOT * F, 256), 256, 0, stream>>>(ei, N_EDGES, E_TOT, hB, ebuf, hA, H, C);
    // mean pool into d_out
    fill_u32<<<cdiv(NGRP * 64, 256), 256, 0, stream>>>((unsigned*)d_out, 0u, NGRP * 64);
    fill_u32<<<1, 64, 0, stream>>>((unsigned*)cnt, 0u, NGRP);
    pool_accum<<<cdiv((long long)N_NODES * F, 256), 256, 0, stream>>>(hA, batch, Bs3, (float*)d_out, cnt, N_NODES, F);
    pool_final<<<cdiv(NGRP * 64, 256), 256, 0, stream>>>((float*)d_out, cnt, NGRP, 64);
  }
}
